// ClusteredLinformerAttention_1176821039316
// MI455X (gfx1250) — matrix operand order repaired and checked by reading, not compiled, not run on hardware
//
#include <hip/hip_runtime.h>
#include <cstdint>
#include <cstddef>

// ---------------------------------------------------------------------------
// Types for CDNA5 WMMA (wave32): v_wmma_f32_16x16x32_bf16
// ---------------------------------------------------------------------------
typedef __bf16 bf16;
typedef __attribute__((ext_vector_type(16))) __bf16 v16bf;
typedef __attribute__((ext_vector_type(8)))  __bf16 v8bf;
typedef __attribute__((ext_vector_type(8)))  float  v8f;

static constexpr int BSZ = 8, NSEQ = 4096, DMODEL = 512, HEADS = 8, RANK = 256, DEPTH = 64;

__device__ __forceinline__ v8f zero8() {
  v8f z;
#pragma unroll
  for (int i = 0; i < 8; ++i) z[i] = 0.0f;
  return z;
}

// ---------------------------------------------------------------------------
// CDNA5 async global->LDS copy (16B per lane), tracked by ASYNCcnt.
//   LDS byte address from the vdst VGPR; global addr from 64-bit VGPR pair.
// ---------------------------------------------------------------------------
__device__ __forceinline__ void async_copy16(uint32_t lds_byte_addr, const void* gaddr) {
  asm volatile("global_load_async_to_lds_b128 %0, %1, off"
               :: "v"(lds_byte_addr), "v"(gaddr)
               : "memory");
}
__device__ __forceinline__ void wait_async0() {
  asm volatile("s_wait_asynccnt 0" ::: "memory");
}
// Generic LDS pointer -> wave-relative LDS byte offset (low 32 bits of aperture addr)
__device__ __forceinline__ uint32_t lds_off(const void* p) {
  return (uint32_t)(uintptr_t)p;
}

// A-matrix 16x32 bf16 fragment (ISA 7.12.2): lane<16 -> K {0..7,16..23},
// lane>=16 -> K {8..15,24..31}; row M = lane&15. `p` points at row start (+kbase).
__device__ __forceinline__ v16bf frag_a(const bf16* p, int lane) {
  const int koff = (lane >> 4) << 3;           // 0 or 8
  v8bf lo = *(const v8bf*)(p + koff);          // 16B -> b128 load
  v8bf hi = *(const v8bf*)(p + koff + 16);
  v16bf r;
#pragma unroll
  for (int i = 0; i < 8; ++i) { r[i] = lo[i]; r[i + 8] = hi[i]; }
  return r;
}

// B-matrix 32x16 bf16 fragment: lane<16 -> K {0..7,16..23}, lane>=16 -> K {8..15,24..31};
// col N = lane&15. LDS must hold B transposed: [n][k] with k contiguous.
__device__ __forceinline__ v16bf frag_b(const bf16* p, int lane) {
  const int koff = (lane >> 4) << 3;           // 0 or 8
  v8bf lo = *(const v8bf*)(p + koff);
  v8bf hi = *(const v8bf*)(p + koff + 16);
  v16bf r;
#pragma unroll
  for (int i = 0; i < 8; ++i) { r[i] = lo[i]; r[i + 8] = hi[i]; }
  return r;
}

__device__ __forceinline__ v8f wmma_bf16(v16bf a, v16bf b, v8f c) {
  // (neg_a, A, neg_b, B, c_mod, C, reuse_a, reuse_b)
  return __builtin_amdgcn_wmma_f32_16x16x32_bf16(false, a, false, b, (short)0, c, false, false);
}

__device__ __forceinline__ float red_max16(float v) {
  v = fmaxf(v, __shfl_xor(v, 1));
  v = fmaxf(v, __shfl_xor(v, 2));
  v = fmaxf(v, __shfl_xor(v, 4));
  v = fmaxf(v, __shfl_xor(v, 8));
  return v;
}
__device__ __forceinline__ float red_sum16(float v) {
  v += __shfl_xor(v, 1);
  v += __shfl_xor(v, 2);
  v += __shfl_xor(v, 4);
  v += __shfl_xor(v, 8);
  return v;
}

// ---------------------------------------------------------------------------
// fp32 -> bf16 conversion
// ---------------------------------------------------------------------------
__global__ void f32_to_bf16(const float* __restrict__ s, bf16* __restrict__ d, size_t n) {
  size_t i = (size_t)blockIdx.x * blockDim.x + threadIdx.x;
  size_t stride = (size_t)gridDim.x * blockDim.x;
  for (; i < n; i += stride) d[i] = (bf16)s[i];
}

// ---------------------------------------------------------------------------
// Generic batched WMMA GEMM: C = alpha * op(A) @ B (+ bias)
//   BM=128, BN=64, BK=32, 256 threads (8 waves); wave w owns rows w*16..w*16+15
//   TRANSA: A stored (K x M) row-major (used for E^T @ k) -> VALU transpose stage
//   !TRANSA: A tile staged with async global->LDS B128 copies (ASYNCcnt path)
//   z batch decomposed as (b, h) with independent strides.
// ---------------------------------------------------------------------------
template <bool TRANSA, bool OUT_BF16, bool BIAS>
__global__ void __launch_bounds__(256)
gemm_wmma(const bf16* __restrict__ A, const bf16* __restrict__ Bm, void* __restrict__ Cv,
          const float* __restrict__ bias, int M, int Ncols, int K,
          int lda, int ldb, int ldc,
          long sAb, long sAh, long sBb, long sBh, long sCb, long sCh,
          int Hdim, float alpha) {
  __shared__ __align__(16) bf16 ldsA[128 * 32];   // [m][k]
  __shared__ __align__(16) bf16 ldsB[64 * 32];    // [n][k] (transposed)

  const int z = blockIdx.z;
  const int bb = z / Hdim, hh = z % Hdim;
  A  += (size_t)bb * sAb + (size_t)hh * sAh;
  Bm += (size_t)bb * sBb + (size_t)hh * sBh;

  const int m0 = blockIdx.x * 128;
  const int n0 = blockIdx.y * 64;
  const int tid = threadIdx.x, lane = tid & 31, wave = tid >> 5;
  const uint32_t ldsA_base = lds_off(ldsA);

  v8f acc[4];
#pragma unroll
  for (int nt = 0; nt < 4; ++nt) acc[nt] = zero8();

  for (int k0 = 0; k0 < K; k0 += 32) {
    if constexpr (TRANSA) {
      for (int i = tid; i < 128 * 32; i += 256) {
        int k = i >> 7, m = i & 127;                       // coalesced along m
        ldsA[m * 32 + k] = A[(size_t)(k0 + k) * lda + (m0 + m)];
      }
    } else {
      // async global->LDS: 128x32 bf16 tile = 512 x 16B chunks, 2 per thread
#pragma unroll
      for (int c = tid; c < 512; c += 256) {
        int m = c >> 2, kc = (c & 3) * 8;                  // 4 chunks per 32-elem row
        async_copy16(ldsA_base + (uint32_t)(m * 32 + kc) * 2,
                     A + (size_t)(m0 + m) * lda + (k0 + kc));
      }
    }
    for (int i = tid; i < 64 * 32; i += 256) {
      int k = i >> 6, n = i & 63;                          // coalesced along n
      ldsB[n * 32 + k] = Bm[(size_t)(k0 + k) * ldb + (n0 + n)];
    }
    if constexpr (!TRANSA) wait_async0();
    __syncthreads();

    v16bf a = frag_a(&ldsA[(wave * 16 + (lane & 15)) * 32], lane);
#pragma unroll
    for (int nt = 0; nt < 4; ++nt) {
      v16bf b = frag_b(&ldsB[(nt * 16 + (lane & 15)) * 32], lane);
      acc[nt] = wmma_bf16(a, b, acc[nt]);
    }
    __syncthreads();
  }

  // Store: C/D layout — lane n = lane&15; VGPR j row = j + 8*(lane>=16)
  const int col0 = n0 + (lane & 15);
  const int row0 = m0 + wave * 16 + ((lane >> 4) << 3);
  float* Cf = (float*)Cv + (size_t)bb * sCb + (size_t)hh * sCh;
  bf16* Cb = (bf16*)Cv + (size_t)bb * sCb + (size_t)hh * sCh;
#pragma unroll
  for (int nt = 0; nt < 4; ++nt) {
#pragma unroll
    for (int j = 0; j < 8; ++j) {
      int row = row0 + j, col = col0 + nt * 16;
      float v = acc[nt][j] * alpha;
      if constexpr (BIAS) v += bias[col];
      if constexpr (OUT_BF16) Cb[(size_t)row * ldc + col] = (bf16)v;
      else                    Cf[(size_t)row * ldc + col] = v;
    }
  }
}

// ---------------------------------------------------------------------------
// Fused Linformer attention per (b, h, 64-row q block).
//   128 threads = 4 waves; wave w owns q rows w*16..w*16+15.
//   Phase 1: S = q @ k_proj^T  (k_proj [r][d] staged via async global->LDS)
//   softmax over R=256 via per-lane tile reduce + 16-lane shfl_xor reduce
//   Phase 2: O = P @ v_proj   (v_proj^T [d][r] restaged into same LDS buffer)
//   Static LDS = 32KB (KV buffer) + 32KB (P tile) = 64KB.
// ---------------------------------------------------------------------------
__global__ void __launch_bounds__(128)
linformer_attn(const bf16* __restrict__ qb, const bf16* __restrict__ kpb,
               const bf16* __restrict__ vpb, bf16* __restrict__ ob) {
  __shared__ __align__(16) bf16 bufKV[RANK * DEPTH];   // phase1: KP[r][d]; phase2: VT[d][r]
  __shared__ __align__(16) bf16 ldsP[64 * RANK];       // P [m][r]

  const int n0 = blockIdx.x * 64;
  const int bh = blockIdx.y, b = bh >> 3, h = bh & 7;
  const int tid = threadIdx.x, lane = tid & 31, wave = tid >> 5;

  const bf16* kp = kpb + (size_t)bh * RANK * DEPTH;
  const bf16* vp = vpb + (size_t)bh * RANK * DEPTH;

  // stage k_proj (row-major copy [r][d]) with async global->LDS B128 copies:
  // 32KB = 2048 x 16B chunks, 16 per thread
  const uint32_t kvbase = lds_off(bufKV);
#pragma unroll
  for (int c = tid; c < 2048; c += 128) {
    async_copy16(kvbase + (uint32_t)c * 16, kp + (size_t)c * 8);
  }
  wait_async0();
  __syncthreads();

  // S = q @ k_proj^T : per wave 16 rows x 256 cols = 16 C tiles, K=64 (2 steps)
  v8f acc[16];
#pragma unroll
  for (int t = 0; t < 16; ++t) acc[t] = zero8();

  const bf16* qrow =
      qb + ((size_t)(b * NSEQ + n0 + wave * 16 + (lane & 15))) * DMODEL + h * DEPTH;
#pragma unroll
  for (int ks = 0; ks < 2; ++ks) {
    v16bf a = frag_a(qrow + ks * 32, lane);   // A fragment straight from global
#pragma unroll
    for (int t = 0; t < 16; ++t) {
      v16bf bfrag = frag_b(&bufKV[(t * 16 + (lane & 15)) * DEPTH + ks * 32], lane);
      acc[t] = wmma_bf16(a, bfrag, acc[t]);
    }
  }

  // softmax over R (scale 1/sqrt(64) = 0.125)
  const float scale = 0.125f;
  float mx[8], sm[8];
#pragma unroll
  for (int j = 0; j < 8; ++j) {
    float m = -1e30f;
#pragma unroll
    for (int t = 0; t < 16; ++t) m = fmaxf(m, acc[t][j] * scale);
    mx[j] = red_max16(m);
  }
#pragma unroll
  for (int j = 0; j < 8; ++j) {
    float s = 0.0f;
#pragma unroll
    for (int t = 0; t < 16; ++t) {
      float e = __expf(acc[t][j] * scale - mx[j]);
      acc[t][j] = e;
      s += e;
    }
    sm[j] = red_sum16(s);
  }

  // write P (bf16) into A-fragment-friendly [m][r] LDS tile
  const int prow0 = wave * 16 + ((lane >> 4) << 3);
#pragma unroll
  for (int t = 0; t < 16; ++t) {
#pragma unroll
    for (int j = 0; j < 8; ++j) {
      ldsP[(prow0 + j) * RANK + t * 16 + (lane & 15)] = (bf16)(acc[t][j] / sm[j]);
    }
  }
  __syncthreads();                                   // all waves done reading bufKV

  // restage v_proj transposed: VT[d][r] (element transpose -> VALU/DS path)
  for (int i = tid; i < RANK * DEPTH; i += 128) {
    int r = i >> 6, d = i & 63;
    bufKV[d * RANK + r] = vp[i];
  }
  __syncthreads();

  // O = P @ v_proj : 16 rows x 64 cols per wave, K=256 (8 steps)
  v8f oacc[4];
#pragma unroll
  for (int nt = 0; nt < 4; ++nt) oacc[nt] = zero8();
#pragma unroll
  for (int ks = 0; ks < 8; ++ks) {
    v16bf a = frag_a(&ldsP[(wave * 16 + (lane & 15)) * RANK + ks * 32], lane);
#pragma unroll
    for (int nt = 0; nt < 4; ++nt) {
      v16bf bfrag = frag_b(&bufKV[(nt * 16 + (lane & 15)) * RANK + ks * 32], lane);
      oacc[nt] = wmma_bf16(a, bfrag, oacc[nt]);
    }
  }

  // store O into concat layout (B, N, D) bf16 at column h*64
  const int row0 = n0 + wave * 16 + ((lane >> 4) << 3);
#pragma unroll
  for (int nt = 0; nt < 4; ++nt) {
#pragma unroll
    for (int j = 0; j < 8; ++j) {
      ob[((size_t)(b * NSEQ + row0 + j)) * DMODEL + h * DEPTH + nt * 16 + (lane & 15)] =
          (bf16)oacc[nt][j];
    }
  }
}

// ---------------------------------------------------------------------------
// Host-side orchestration
// ---------------------------------------------------------------------------
extern "C" void kernel_launch(void* const* d_in, const int* in_sizes, int n_in,
                              void* d_out, int out_size, void* d_ws, size_t ws_size,
                              hipStream_t stream) {
  const float* x  = (const float*)d_in[0];
  const float* wq = (const float*)d_in[1];
  const float* wk = (const float*)d_in[2];
  const float* wv = (const float*)d_in[3];
  const float* E  = (const float*)d_in[4];
  const float* F  = (const float*)d_in[5];
  const float* wd = (const float*)d_in[6];
  const float* bd = (const float*)d_in[7];
  float* out = (float*)d_out;

  const size_t XE = (size_t)BSZ * NSEQ * DMODEL;        // 16,777,216
  const size_t WE = (size_t)DMODEL * DMODEL;            // 262,144
  const size_t EE = (size_t)HEADS * NSEQ * RANK;        // 8,388,608
  const size_t PE = (size_t)BSZ * HEADS * RANK * DEPTH; // 1,048,576

  bf16* p = (bf16*)d_ws;
  bf16 *xb = p;  p += XE;
  bf16 *qb = p;  p += XE;
  bf16 *kb = p;  p += XE;
  bf16 *vb = p;  p += XE;
  bf16 *ob = p;  p += XE;
  bf16 *wqb = p; p += WE;
  bf16 *wkb = p; p += WE;
  bf16 *wvb = p; p += WE;
  bf16 *wdb = p; p += WE;
  bf16 *Eb = p;  p += EE;
  bf16 *Fb = p;  p += EE;
  bf16 *kpb = p; p += PE;
  bf16 *vpb = p; p += PE;

  auto cvt = [&](const float* s, bf16* d, size_t n) {
    int blocks = (int)((n + 256 * 8 - 1) / (256 * 8));
    f32_to_bf16<<<blocks, 256, 0, stream>>>(s, d, n);
  };
  cvt(x, xb, XE);
  cvt(wq, wqb, WE); cvt(wk, wkb, WE); cvt(wv, wvb, WE); cvt(wd, wdb, WE);
  cvt(E, Eb, EE);  cvt(F, Fb, EE);

  const dim3 blk(256);
  const int M = BSZ * NSEQ;                       // 32768

  // QKV projections: (32768 x 512) @ (512 x 512) -> bf16
  dim3 g1(M / 128, DMODEL / 64, 1);
  gemm_wmma<false, true, false><<<g1, blk, 0, stream>>>(
      xb, wqb, qb, nullptr, M, DMODEL, DMODEL, DMODEL, DMODEL, DMODEL,
      0, 0, 0, 0, 0, 0, 1, 1.0f);
  gemm_wmma<false, true, false><<<g1, blk, 0, stream>>>(
      xb, wkb, kb, nullptr, M, DMODEL, DMODEL, DMODEL, DMODEL, DMODEL,
      0, 0, 0, 0, 0, 0, 1, 1.0f);
  gemm_wmma<false, true, false><<<g1, blk, 0, stream>>>(
      xb, wvb, vb, nullptr, M, DMODEL, DMODEL, DMODEL, DMODEL, DMODEL,
      0, 0, 0, 0, 0, 0, 1, 1.0f);

  // Linformer projections: k_proj[b,h] = E[h]^T (256x4096) @ k[b,:,h*64..] (4096x64)
  dim3 g2(RANK / 128, DEPTH / 64, BSZ * HEADS);
  gemm_wmma<true, true, false><<<g2, blk, 0, stream>>>(
      Eb, kb, kpb, nullptr, RANK, DEPTH, NSEQ, RANK, DMODEL, DEPTH,
      0, (long)NSEQ * RANK,                       // A: per-head E
      (long)NSEQ * DMODEL, (long)DEPTH,           // B: per-b rows, per-h col slice
      (long)HEADS * RANK * DEPTH, (long)RANK * DEPTH,  // C: (bh, R, 64)
      HEADS, 1.0f);
  gemm_wmma<true, true, false><<<g2, blk, 0, stream>>>(
      Fb, vb, vpb, nullptr, RANK, DEPTH, NSEQ, RANK, DMODEL, DEPTH,
      0, (long)NSEQ * RANK,
      (long)NSEQ * DMODEL, (long)DEPTH,
      (long)HEADS * RANK * DEPTH, (long)RANK * DEPTH,
      HEADS, 1.0f);

  // Fused attention: scores + softmax + weighted sum of v_proj
  dim3 ga(NSEQ / 64, BSZ * HEADS, 1);
  linformer_attn<<<ga, dim3(128), 0, stream>>>(qb, kpb, vpb, ob);

  // Final dense: (32768 x 512) @ w_dense + b_dense -> fp32 out
  gemm_wmma<false, false, true><<<g1, blk, 0, stream>>>(
      ob, wdb, out, bd, M, DMODEL, DMODEL, DMODEL, DMODEL, DMODEL,
      0, 0, 0, 0, 0, 0, 1, 1.0f);
}
